// GroupedQueryAttention_76819785056413
// MI455X (gfx1250) — matrix operand order repaired and checked by reading, not compiled, not run on hardware
//
#include <hip/hip_runtime.h>
#include <math.h>

// ---------------------------------------------------------------- types
typedef __bf16 bf16;
typedef bf16  v16bf __attribute__((ext_vector_type(16)));
typedef bf16  v8bf  __attribute__((ext_vector_type(8)));
typedef float v8f   __attribute__((ext_vector_type(8)));

#define WMMA_BF16(a, b, c) \
  __builtin_amdgcn_wmma_f32_16x16x32_bf16(false, (a), false, (b), (short)0, (c), false, false)

// ---------------------------------------------------------------- problem constants
constexpr int BATCH = 2;
constexpr int T     = 2048;
constexpr int DMODEL= 2048;
constexpr int H     = 32;
constexpr int KVH   = 8;
constexpr int HD    = 64;
constexpr int MROWS = BATCH * T;           // 4096
constexpr int NKV   = 2 * KVH * HD;        // 1024
constexpr float SCALE = 0.125f;            // HD^-0.5

__device__ inline v16bf load16(const bf16* p) {
  v16bf r;
  *reinterpret_cast<v8bf*>(&r)       = *reinterpret_cast<const v8bf*>(p);
  *(reinterpret_cast<v8bf*>(&r) + 1) = *reinterpret_cast<const v8bf*>(p + 8);
  return r;
}
__device__ inline v16bf load8x2(const bf16* p0, const bf16* p1) {
  v16bf r;
  *reinterpret_cast<v8bf*>(&r)       = *reinterpret_cast<const v8bf*>(p0);
  *(reinterpret_cast<v8bf*>(&r) + 1) = *reinterpret_cast<const v8bf*>(p1);
  return r;
}

// ---------------------------------------------------------------- fp32 -> bf16 convert
__global__ __launch_bounds__(256) void cvt_bf16(const float* __restrict__ in,
                                                bf16* __restrict__ out, long n) {
  long i = (long)blockIdx.x * 256 + threadIdx.x;
  if (i < n) out[i] = (bf16)in[i];
}

// fp32 [R,C] -> bf16 transposed [C,R]
__global__ __launch_bounds__(256) void cvt_t_bf16(const float* __restrict__ in,
                                                  bf16* __restrict__ out, int R, int C) {
  long i = (long)blockIdx.x * 256 + threadIdx.x;
  if (i < (long)R * C) {
    int r = (int)(i / C);
    int c = (int)(i % C);
    out[(size_t)c * R + r] = (bf16)in[i];
  }
}

// ---------------------------------------------------------------- generic bf16 WMMA GEMM
// C[M,N] = A[M,K] * BT[N,K]^T ; wave computes 64x64, block = 4 waves = 128x128 tile.
// mode 0: bf16 out0[M*N]          (Q projection)
// mode 1: split KV store: n<512 -> bf16 k[M*512]; n>=512 -> bf16 v^T[(kvh*64+d)*T + t]
// mode 2: f32 out0[M*N]           (final output projection)
__global__ __launch_bounds__(128) void gemm_wmma(const bf16* __restrict__ A,
                                                 const bf16* __restrict__ BT,
                                                 void* __restrict__ out0,
                                                 void* __restrict__ out1,
                                                 int M, int N, int K, int mode) {
  const int lane = threadIdx.x & 31;
  const int wave = threadIdx.x >> 5;
  const int half = lane >> 4;
  const int l16  = lane & 15;
  const int m0 = blockIdx.y * 128 + (wave >> 1) * 64;
  const int n0 = blockIdx.x * 128 + (wave & 1) * 64;

  const bf16* aptr[4];
  const bf16* bptr[4];
#pragma unroll
  for (int i = 0; i < 4; ++i)
    aptr[i] = A + (size_t)(m0 + i * 16 + l16) * K + half * 16;
#pragma unroll
  for (int j = 0; j < 4; ++j)
    bptr[j] = BT + (size_t)(n0 + j * 16 + l16) * K + half * 16;

  v8f acc[4][4] = {};

  for (int k = 0; k < K; k += 32) {
    v16bf a[4], b[4];
#pragma unroll
    for (int i = 0; i < 4; ++i)
      a[i] = load16(aptr[i] + k);
#pragma unroll
    for (int j = 0; j < 4; ++j)
      b[j] = load16(bptr[j] + k);
    // prefetch next-next K slab (lowers to global_prefetch_b8)
    __builtin_prefetch((const void*)(aptr[0] + k + 64), 0, 1);
    __builtin_prefetch((const void*)(bptr[0] + k + 64), 0, 1);
#pragma unroll
    for (int i = 0; i < 4; ++i)
#pragma unroll
      for (int j = 0; j < 4; ++j)
        acc[i][j] = WMMA_BF16(a[i], b[j], acc[i][j]);
  }

  // epilogue
#pragma unroll
  for (int i = 0; i < 4; ++i) {
#pragma unroll
    for (int j = 0; j < 4; ++j) {
#pragma unroll
      for (int r = 0; r < 8; ++r) {
        const int m = m0 + i * 16 + r + 8 * half;
        const int n = n0 + j * 16 + l16;
        const float v = acc[i][j][r];
        if (mode == 0) {
          ((bf16*)out0)[(size_t)m * N + n] = (bf16)v;
        } else if (mode == 2) {
          ((float*)out0)[(size_t)m * N + n] = v;
        } else { // mode 1: split K / V^T
          if (n < 512) {
            ((bf16*)out0)[(size_t)m * 512 + n] = (bf16)v;
          } else {
            const int n2  = n - 512;
            const int kvh = n2 >> 6;
            const int d   = n2 & 63;
            const int bb  = m >> 11;   // /T
            const int t   = m & 2047;  // %T
            ((bf16*)out1)[(((size_t)(bb * KVH + kvh)) * HD + d) * T + t] = (bf16)v;
          }
        }
      }
    }
  }
}

// ---------------------------------------------------------------- causal flash attention
// q   : bf16 [B, T, H*HD]   (row-major, col = h*64+d)
// k   : bf16 [B, T, KVH*HD]
// vt  : bf16 [B, KVH, HD, T]   (transposed V)
// ao  : bf16 [B, T, H*HD]
// block = 4 waves; wave handles 16 query rows; grid (T/64, H, B)
__global__ __launch_bounds__(128) void flash_attn(const bf16* __restrict__ q,
                                                  const bf16* __restrict__ k,
                                                  const bf16* __restrict__ vt,
                                                  bf16* __restrict__ ao) {
  __shared__ float pbuf[4][16][32];

  const int lane = threadIdx.x & 31;
  const int wave = threadIdx.x >> 5;
  const int half = lane >> 4;
  const int l16  = lane & 15;

  const int b   = blockIdx.z;
  const int h   = blockIdx.y;
  const int kvh = h >> 2;               // GROUPS = H/KVH = 4 (repeat_interleave)
  const int q0  = blockIdx.x * 64 + wave * 16;

  // Q A-fragments over head-dim (K=64 -> 2 frags of K=32)
  const bf16* qrow = q + ((size_t)b * T + q0 + l16) * DMODEL + (size_t)h * HD;
  v16bf qa[2];
#pragma unroll
  for (int s = 0; s < 2; ++s)
    qa[s] = load16(qrow + s * 32 + half * 16);

  const bf16* kbase = k  + (size_t)b * T * (KVH * HD) + (size_t)kvh * HD;
  const bf16* vbase = vt + ((size_t)(b * KVH + kvh) * HD) * T;

  float mrow[8], lrow[8];
  v8f oacc[4] = {};
#pragma unroll
  for (int r = 0; r < 8; ++r) { mrow[r] = -INFINITY; lrow[r] = 0.0f; }

  const int kend = q0 + 16;             // causal: keys < q_row+1
  for (int kb = 0; kb < kend; kb += 32) {
    // ---- S = Q K^T  (two 16-key tiles, contraction over 64 dims)
    v8f s[2] = {};
#pragma unroll
    for (int t = 0; t < 2; ++t) {
      const bf16* krow = kbase + (size_t)(kb + t * 16 + l16) * (KVH * HD) + half * 16;
      v16bf kb0 = load16(krow);        // dims 0..31
      v16bf kb1 = load16(krow + 32);   // dims 32..63
      s[t] = WMMA_BF16(qa[0], kb0, s[t]);
      s[t] = WMMA_BF16(qa[1], kb1, s[t]);
    }

    // ---- mask + online softmax update (row lives on one 16-lane group)
#pragma unroll
    for (int r = 0; r < 8; ++r) {
      const int row = q0 + r + 8 * half;
      float v0 = s[0][r] * SCALE;
      float v1 = s[1][r] * SCALE;
      if (kb + l16 > row)      v0 = -INFINITY;
      if (kb + 16 + l16 > row) v1 = -INFINITY;
      float mx = fmaxf(v0, v1);
#pragma unroll
      for (int msk = 1; msk <= 8; msk <<= 1)
        mx = fmaxf(mx, __shfl_xor(mx, msk, 32));
      const float mnew = fmaxf(mrow[r], mx);
      const float c = __expf(mrow[r] - mnew);
      const float p0 = __expf(v0 - mnew);
      const float p1 = __expf(v1 - mnew);
      float sum = p0 + p1;
#pragma unroll
      for (int msk = 1; msk <= 8; msk <<= 1)
        sum += __shfl_xor(sum, msk, 32);
      lrow[r] = lrow[r] * c + sum;
      mrow[r] = mnew;
#pragma unroll
      for (int j = 0; j < 4; ++j) oacc[j][r] *= c;
      // spill P (C-layout) to LDS for A-fragment re-layout
      pbuf[wave][r + 8 * half][l16]      = p0;
      pbuf[wave][r + 8 * half][16 + l16] = p1;
    }
    asm volatile("s_wait_dscnt 0" ::: "memory");   // wave-local LDS RAW ordering

    // ---- re-read P as 16x32 bf16 A-fragment
    v16bf pa;
    {
      const float* prow = &pbuf[wave][l16][0];
      const int koff = half * 16;
#pragma unroll
      for (int e = 0; e < 8; ++e) pa[e]     = (bf16)prow[koff + e];
#pragma unroll
      for (int e = 0; e < 8; ++e) pa[8 + e] = (bf16)prow[koff + 8 + e];
    }
    asm volatile("s_wait_dscnt 0" ::: "memory");

    // ---- O += P V  (V^T rows are contiguous in keys)
#pragma unroll
    for (int j = 0; j < 4; ++j) {
      const bf16* vrow = vbase + (size_t)(j * 16 + l16) * T + kb + half * 16;
      v16bf vb = load16(vrow);
      oacc[j] = WMMA_BF16(pa, vb, oacc[j]);
    }
  }

  // ---- normalize + store
  bf16* aop = ao + ((size_t)b * T + q0) * DMODEL + (size_t)h * HD;
#pragma unroll
  for (int r = 0; r < 8; ++r) {
    const float inv = 1.0f / lrow[r];
    const int rowt = r + 8 * half;
#pragma unroll
    for (int j = 0; j < 4; ++j)
      aop[(size_t)rowt * DMODEL + j * 16 + l16] = (bf16)(oacc[j][r] * inv);
  }
}

// ---------------------------------------------------------------- launch
extern "C" void kernel_launch(void* const* d_in, const int* in_sizes, int n_in,
                              void* d_out, int out_size, void* d_ws, size_t ws_size,
                              hipStream_t stream) {
  (void)in_sizes; (void)n_in; (void)out_size; (void)ws_size;
  const float* x   = (const float*)d_in[0];
  const float* wq  = (const float*)d_in[1];
  const float* wkv = (const float*)d_in[2];
  const float* wo  = (const float*)d_in[3];

  char* ws = (char*)d_ws;
  size_t off = 0;
  bf16* x_bf   = (bf16*)(ws + off); off += (size_t)MROWS * DMODEL * 2;   // 16 MiB
  bf16* wqT    = (bf16*)(ws + off); off += (size_t)DMODEL * DMODEL * 2;  //  8 MiB
  bf16* wkvT   = (bf16*)(ws + off); off += (size_t)NKV * DMODEL * 2;     //  4 MiB
  bf16* woT    = (bf16*)(ws + off); off += (size_t)DMODEL * DMODEL * 2;  //  8 MiB
  bf16* q_ws   = (bf16*)(ws + off); off += (size_t)MROWS * DMODEL * 2;   // 16 MiB
  bf16* k_ws   = (bf16*)(ws + off); off += (size_t)MROWS * 512 * 2;      //  4 MiB
  bf16* v_ws   = (bf16*)(ws + off); off += (size_t)MROWS * 512 * 2;      //  4 MiB
  bf16* ao_ws  = (bf16*)(ws + off); off += (size_t)MROWS * DMODEL * 2;   // 16 MiB

  // 1) convert / transpose to bf16
  {
    long n = (long)MROWS * DMODEL;
    cvt_bf16<<<(unsigned)((n + 255) / 256), 256, 0, stream>>>(x, x_bf, n);
  }
  {
    long n = (long)DMODEL * DMODEL;
    cvt_t_bf16<<<(unsigned)((n + 255) / 256), 256, 0, stream>>>(wq, wqT, DMODEL, DMODEL);
    cvt_t_bf16<<<(unsigned)((n + 255) / 256), 256, 0, stream>>>(wo, woT, DMODEL, DMODEL);
  }
  {
    long n = (long)DMODEL * NKV;
    cvt_t_bf16<<<(unsigned)((n + 255) / 256), 256, 0, stream>>>(wkv, wkvT, DMODEL, NKV);
  }

  // 2) Q projection: [4096,2048] x [2048,2048]
  gemm_wmma<<<dim3(DMODEL / 128, MROWS / 128), 128, 0, stream>>>(
      x_bf, wqT, (void*)q_ws, nullptr, MROWS, DMODEL, DMODEL, 0);

  // 3) KV projection: [4096,2048] x [2048,1024] -> k + v^T
  gemm_wmma<<<dim3(NKV / 128, MROWS / 128), 128, 0, stream>>>(
      x_bf, wkvT, (void*)k_ws, (void*)v_ws, MROWS, NKV, DMODEL, 1);

  // 4) causal flash attention
  flash_attn<<<dim3(T / 64, H, BATCH), 128, 0, stream>>>(q_ws, k_ws, v_ws, ao_ws);

  // 5) output projection -> fp32 d_out
  gemm_wmma<<<dim3(DMODEL / 128, MROWS / 128), 128, 0, stream>>>(
      ao_ws, woT, d_out, nullptr, MROWS, DMODEL, DMODEL, 2);
}
